// VQ_39754217291940
// MI455X (gfx1250) — compile-verified, matrix operand order checked by reading, not run on hardware
//
#include <hip/hip_runtime.h>

typedef __attribute__((ext_vector_type(16))) _Float16 v16h;
typedef __attribute__((ext_vector_type(8)))  float    v8f;

#define N_EMB   1024
#define EMB_DIM 256
#define NPIX    65536  // 64 * 32 * 32

// --------------------------------------------------------------------------
// Kernel 1: codebook prep.
//  - exact f32 squared norms e2[n]
//  - split-f16 (hi + lo) codebook, pre-swizzled into the CDNA5 WMMA
//    B-fragment layout for V_WMMA_F32_16X16X32_F16 so the main kernel can
//    fetch a lane's whole fragment as two contiguous b128 loads.
//    Fragment addressing: tile t = n/16, kstep s = k/32,
//    lane = (n%16) + 16*hh, half j with K(j,hh) = (j>>3)*16 + hh*8 + (j&7).
// --------------------------------------------------------------------------
__global__ __launch_bounds__(256) void vq_prep(const float* __restrict__ emb,
                                               _Float16* __restrict__ hi,
                                               _Float16* __restrict__ lo,
                                               float* __restrict__ e2) {
  const int n = blockIdx.x;   // code row 0..1023
  const int k = threadIdx.x;  // dim 0..255
  const float v = emb[n * EMB_DIM + k];

  __shared__ float red[256];
  red[k] = v * v;
  __syncthreads();
  for (int off = 128; off > 0; off >>= 1) {
    if (k < off) red[k] += red[k + off];
    __syncthreads();
  }
  if (k == 0) e2[n] = red[0];

  const _Float16 h = (_Float16)v;
  const _Float16 l = (_Float16)(v - (float)h);

  const int t    = n >> 4;
  const int s    = k >> 5;
  const int r    = k & 31;
  const int part = r >> 4;   // 0/1 -> fragment halves j<8 / j>=8
  const int rem  = r & 15;
  const int hh   = rem >> 3; // lane half
  const int jj   = rem & 7;
  const int j    = part * 8 + jj;
  const int lane = (n & 15) + (hh << 4);
  const size_t addr = ((size_t)(t * 8 + s) * 32 + lane) * 16 + j;
  hi[addr] = h;
  lo[addr] = l;
}

// --------------------------------------------------------------------------
// Kernel 2: fused distance GEMM + argmin.
// Block = 256 threads = 8 waves; each wave owns 16 pixels (block: 128).
// score[p][n] = e2[n] - 2 * dot(z_p, e_n)   (||z||^2 dropped: argmin-invariant)
// dot computed as 3 f16 WMMAs per 32-wide K step (split-f16 ~= fp32 accuracy).
// B tiles are register-double-buffered: global loads for tile t+1 are issued
// before the 24 WMMAs of tile t, so L2 latency overlaps matrix compute.
// --------------------------------------------------------------------------
__global__ __launch_bounds__(256) void vq_argmin(const float* __restrict__ z,
                                                 const v16h* __restrict__ hiV,
                                                 const v16h* __restrict__ loV,
                                                 const float* __restrict__ e2,
                                                 int* __restrict__ out_idx) {
  __shared__ v16h lds_hi[8][32];
  __shared__ v16h lds_lo[8][32];

  const int tid   = threadIdx.x;
  const int w     = tid >> 5;   // wave in block
  const int L     = tid & 31;   // lane
  const int m     = L & 15;     // A-fragment row (pixel within wave tile)
  const int hh    = L >> 4;     // lane half -> K offset
  const int pbase = blockIdx.x * 128 + w * 16;
  const int p     = pbase + m;  // this lane's pixel for A loads
  // flat_z[p][k] lives at z[(p>>10)*262144 + k*1024 + (p&1023)]
  const float* zb = z + ((size_t)(p >> 10) * 262144 + (size_t)(p & 1023));

  // Build A fragments (hi + lo) for all 8 K-steps, ISA 16-bit A layout.
  v16h ah[8], al[8];
#pragma unroll
  for (int s = 0; s < 8; ++s) {
#pragma unroll
    for (int j = 0; j < 16; ++j) {
      const int k = (s << 5) + ((j >> 3) << 4) + (hh << 3) + (j & 7);
      const float v = zb[(size_t)k << 10];
      const _Float16 hv = (_Float16)v;
      ah[s][j] = hv;
      al[s][j] = (_Float16)(v - (float)hv);
    }
  }

  float best[8];
  int   bidx[8];
#pragma unroll
  for (int r2 = 0; r2 < 8; ++r2) { best[r2] = 3.4e38f; bidx[r2] = 0; }

  // Each thread stages one v16h of hi and one of lo per tile.
  const int cs = tid >> 5;
  const int cl = tid & 31;
  const size_t coff = (size_t)cs * 32 + cl;

  // Prologue: fetch tile 0 into registers.
  v16h reg_hi = hiV[coff];
  v16h reg_lo = loV[coff];

#pragma unroll 1
  for (int t = 0; t < 64; ++t) {  // 64 code tiles of 16
    __syncthreads();              // all waves done reading previous tile
    lds_hi[cs][cl] = reg_hi;
    lds_lo[cs][cl] = reg_lo;
    __syncthreads();              // tile t visible to all waves

    // Issue global loads for tile t+1 now; they complete during the WMMAs.
    if (t < 63) {
      const size_t noff = (size_t)(t + 1) * 256 + coff;
      reg_hi = hiV[noff];
      reg_lo = loV[noff];
    }

    v8f acc0 = {};  // hi*hi
    v8f acc1 = {};  // cross terms
#pragma unroll
    for (int s = 0; s < 8; ++s) {
      const v16h bh = lds_hi[s][L];
      const v16h bl = lds_lo[s][L];
      acc1 = __builtin_amdgcn_wmma_f32_16x16x32_f16(false, al[s], false, bh,
                                                    (short)0, acc1, false, false);
      acc1 = __builtin_amdgcn_wmma_f32_16x16x32_f16(false, ah[s], false, bl,
                                                    (short)0, acc1, false, false);
      acc0 = __builtin_amdgcn_wmma_f32_16x16x32_f16(false, ah[s], false, bh,
                                                    (short)0, acc0, false, false);
    }

    // C layout: lane L holds N = L%16 = m, M = hh*8 + r2 (VGPR r2)
    const int   n   = (t << 4) + m;
    const float e2n = e2[n];
#pragma unroll
    for (int r2 = 0; r2 < 8; ++r2) {
      const float sc = fmaf(-2.0f, acc0[r2] + acc1[r2], e2n);
      if (sc < best[r2]) { best[r2] = sc; bidx[r2] = n; }  // strict <: first min
    }
  }

  // Min-reduce across the 16 lanes holding the same pixel row (xor masks 1..8
  // stay inside each 16-lane half). Tie-break: smaller code index (jnp.argmin).
#pragma unroll
  for (int r2 = 0; r2 < 8; ++r2) {
#pragma unroll
    for (int mask = 1; mask < 16; mask <<= 1) {
      const float os = __shfl_xor(best[r2], mask, 32);
      const int   oi = __shfl_xor(bidx[r2], mask, 32);
      if (os < best[r2] || (os == best[r2] && oi < bidx[r2])) {
        best[r2] = os; bidx[r2] = oi;
      }
    }
  }
  if (m == 0) {
#pragma unroll
    for (int r2 = 0; r2 < 8; ++r2)
      out_idx[pbase + (hh << 3) + r2] = bidx[r2];
  }
}

// --------------------------------------------------------------------------
// Kernel 3: z_q gather. Block handles 64 consecutive pixels; thread (t&63) =
// pixel, (t>>6) = 64-wide d-chunk. Stores are wave-contiguous along hw;
// codebook reads are float4 L2 hits.
// --------------------------------------------------------------------------
__global__ __launch_bounds__(256) void vq_gather(const float4* __restrict__ emb4,
                                                 const int* __restrict__ idx,
                                                 float* __restrict__ zq) {
  const int tid  = threadIdx.x;
  const int pix  = blockIdx.x * 64 + (tid & 63);
  const int dc   = tid >> 6;  // 0..3
  const int code = idx[pix];
  const size_t obase = (size_t)(pix >> 10) * 262144 + (size_t)(pix & 1023);
  const float4* row = emb4 + (size_t)code * 64 + (size_t)dc * 16;
  float* ob = zq + obase + (size_t)dc * 64 * 1024;
#pragma unroll
  for (int q = 0; q < 16; ++q) {
    const float4 v = row[q];
    ob[(size_t)(q * 4 + 0) * 1024] = v.x;
    ob[(size_t)(q * 4 + 1) * 1024] = v.y;
    ob[(size_t)(q * 4 + 2) * 1024] = v.z;
    ob[(size_t)(q * 4 + 3) * 1024] = v.w;
  }
}

// --------------------------------------------------------------------------
extern "C" void kernel_launch(void* const* d_in, const int* in_sizes, int n_in,
                              void* d_out, int out_size, void* d_ws, size_t ws_size,
                              hipStream_t stream) {
  const float* z   = (const float*)d_in[0];  // [64,256,32,32]
  const float* emb = (const float*)d_in[1];  // [1024,256]

  // workspace: hi f16 (512KB) | lo f16 (512KB) | e2 f32 (4KB)
  _Float16* ws_hi = (_Float16*)d_ws;
  _Float16* ws_lo = ws_hi + (size_t)N_EMB * EMB_DIM;
  float*    ws_e2 = (float*)(ws_lo + (size_t)N_EMB * EMB_DIM);

  int*   out_idx = (int*)d_out;             // 65536 int32 indices
  float* zq      = (float*)d_out + NPIX;    // [64,256,32,32] f32

  vq_prep<<<dim3(N_EMB), dim3(256), 0, stream>>>(emb, ws_hi, ws_lo, ws_e2);
  vq_argmin<<<dim3(NPIX / 128), dim3(256), 0, stream>>>(
      z, (const v16h*)ws_hi, (const v16h*)ws_lo, ws_e2, out_idx);
  vq_gather<<<dim3(NPIX / 64), dim3(256), 0, stream>>>(
      (const float4*)emb, out_idx, zq);
}
